// TextLSTM_25546465477017
// MI455X (gfx1250) — matrix operand order repaired and checked by reading, not compiled, not run on hardware
//
#include <hip/hip_runtime.h>
#include <hip/hip_bf16.h>

// MI455X (gfx1250) implementation.
// Phase 0: one-time conversion of W_ih -> f16 B-panel in WMMA LDS layout.
// Phase 1: x_proj GEMM with v_wmma_f32_16x16x32_f16; B panel pulled into LDS
//          with GLOBAL_LOAD_ASYNC_TO_LDS_B128 (ASYNCcnt path, no VGPR trip);
//          A/B fragments are 2x ds_load_b128 each (K contiguous: k = 16*half + s).
// Phase 2: persistent single-workgroup LSTM scan (latency-bound, dominates),
//          W_hh in VGPRs, h broadcast via float4 LDS reads, x_proj prefetched
//          8 steps ahead in a register ring. x_proj stored f16 (160MB, fits
//          the 192MB L2 so the sequential consumer hits L2, not HBM).
// Phase 3: classifier folded into scan kernel tail.

typedef __attribute__((ext_vector_type(16))) _Float16 v16h;
typedef __attribute__((ext_vector_type(8)))  _Float16 v8h;
typedef __attribute__((ext_vector_type(8)))  float    v8f;

#define H 100
#define G4 400    // 4*H
#define KPAD 128  // K padded to multiple of 32
#define NCHUNK 4  // KPAD/32
#define ASTRIDE 136  // 272 B row stride: 16B-aligned, bank-conflict-free
#define BSTRIDE 40   // 80 B row stride: 16B-aligned, bank-conflict-free

#define BPANEL_HALVES (NCHUNK * G4 * BSTRIDE)       // 64000 halves
#define BPANEL_BYTES  (BPANEL_HALVES * 2)           // 128000 B
#define AS_BYTES      (32 * ASTRIDE * 2)            // 8704 B
#define BIAS_BYTES    (G4 * 4)                      // 1600 B
#define SMEM_BYTES    (BPANEL_BYTES + AS_BYTES + BIAS_BYTES)

union AF16 { v16h v; v8h p[2]; };
union PK2 { _Float16 h[2]; unsigned u; };

// ---------------------------------------------------------------------------
// Kernel 0: build f16 W_ih^T panel in the exact LDS layout: Bp[kc][n][kk]
// ---------------------------------------------------------------------------
__global__ __launch_bounds__(256) void prep_bpanel_kernel(
    const float* __restrict__ W_ih, _Float16* __restrict__ Bp)
{
    for (int idx = threadIdx.x; idx < NCHUNK * G4 * (BSTRIDE / 2); idx += 256) {
        int kc  = idx / (G4 * (BSTRIDE / 2));
        int rem = idx % (G4 * (BSTRIDE / 2));
        int n   = rem / (BSTRIDE / 2);
        int kk  = (rem % (BSTRIDE / 2)) * 2;
        int kg  = kc * 32 + kk;
        float v0 = (kk     < 32 && kg     < H) ? W_ih[n * H + kg]     : 0.f;
        float v1 = (kk + 1 < 32 && kg + 1 < H) ? W_ih[n * H + kg + 1] : 0.f;
        PK2 pk; pk.h[0] = (_Float16)v0; pk.h[1] = (_Float16)v1;
        *(unsigned*)&Bp[(size_t)kc * G4 * BSTRIDE + n * BSTRIDE + kk] = pk.u;
    }
}

// ---------------------------------------------------------------------------
// Kernel A: xp[t][j] = sum_k x[t,k] * W_ih[j,k] + (b_ih[j] + b_hh[j]), f16 out
// Block: 256 threads (8 waves), 32 rows of x, all 400 output columns.
// Waves 0-3 own M-tile 0, waves 4-7 own M-tile 1 (A fragment fixed per wave).
// ---------------------------------------------------------------------------
__global__ __launch_bounds__(256) void xproj_wmma_kernel(
    const float* __restrict__ x, const _Float16* __restrict__ Bp,
    const float* __restrict__ b_ih, const float* __restrict__ b_hh,
    _Float16* __restrict__ xp, int T)
{
    extern __shared__ __attribute__((aligned(16))) char smem[];
    // B panel (all 4 K-chunks): Bpan_s[kc][n][kk], K contiguous per column
    _Float16 (*Bpan_s)[G4][BSTRIDE] =
        (_Float16 (*)[G4][BSTRIDE])smem;
    // A tile: K contiguous per 32-chunk so a fragment is 16 contiguous halves:
    //   chunk layout [K0..31]
    _Float16 (*As)[ASTRIDE] = (_Float16 (*)[ASTRIDE])(smem + BPANEL_BYTES);
    float* bias_s = (float*)(smem + BPANEL_BYTES + AS_BYTES);

    const int tid     = threadIdx.x;
    const int lane    = tid & 31;
    const int wave    = tid >> 5;
    const int mt      = wave >> 2;     // 0 or 1: this wave's M-tile
    const int nq      = wave & 3;      // N-tile phase
    const int rowBase = blockIdx.x * 32;

    // --- Async copy of the whole B panel: global -> LDS, no VGPR round trip.
    // 250 wave-issues of global_load_async_to_lds_b128 (512 B per issue).
    {
        unsigned ldsBase = (unsigned)(size_t)&Bpan_s[0][0][0];
        unsigned long long gBase = (unsigned long long)(size_t)Bp;
        for (int off = tid * 16; off < BPANEL_BYTES; off += 256 * 16) {
            unsigned la = ldsBase + (unsigned)off;
            unsigned long long ga = gBase + (unsigned long long)off;
            asm volatile("global_load_async_to_lds_b128 %0, %1, off"
                         :: "v"(la), "v"(ga) : "memory");
        }
        asm volatile("s_wait_asynccnt 0x0" ::: "memory");
    }

    // Stage A tile (fp32 -> f16 pairs, K-contiguous, zero-pad, row-guarded)
    for (int idx = tid; idx < 32 * (KPAD / 2); idx += 256) {
        int r = idx >> 6;
        int k = (idx & 63) * 2;        // even, pair stays in same 8-block
        int grow = rowBase + r;
        float v0 = (k     < H && grow < T) ? x[(long)grow * H + k]     : 0.f;
        float v1 = (k + 1 < H && grow < T) ? x[(long)grow * H + k + 1] : 0.f;
        int chunk = k >> 5, kk = k & 31, bb = kk >> 3;
        int bbmap = ((bb & 1) << 1) | (bb >> 1);   // 0,1,2,3 -> 0,2,1,3
        int pos = chunk * 32 + kk;
        PK2 pk; pk.h[0] = (_Float16)v0; pk.h[1] = (_Float16)v1;
        *(unsigned*)&As[r][pos] = pk.u;
    }
    for (int j = tid; j < G4; j += 256) bias_s[j] = b_ih[j] + b_hh[j];
    __syncthreads();   // LDS fully staged (async panel + A tile + bias)

    v8f acc[7];
    v8f vzero = {0.f, 0.f, 0.f, 0.f, 0.f, 0.f, 0.f, 0.f};
#pragma unroll
    for (int i = 0; i < 7; ++i) acc[i] = vzero;

    const int r0  = lane & 15;
    const int sel = lane >> 4;         // 0: low half-wave, 1: high half-wave

    // All 28 WMMAs in a straight line: no restaging, no barriers.
#pragma unroll
    for (int kc = 0; kc < NCHUNK; ++kc) {
        AF16 a;   // 2x ds_load_b128 (16 K-contiguous halves)
        {
            const v8h* pa = (const v8h*)&As[mt * 16 + r0][kc * 32 + sel * 16];
            a.p[0] = pa[0];
            a.p[1] = pa[1];
        }
#pragma unroll
        for (int ti = 0; ti < 7; ++ti) {
            int nt = nq + ti * 4;
            if (nt >= 25) continue;
            AF16 b;   // 2x ds_load_b128 (K contiguous per column)
            const v8h* pb = (const v8h*)&Bpan_s[kc][nt * 16 + r0][sel * 16];
            b.p[0] = pb[0];
            b.p[1] = pb[1];
            acc[ti] = __builtin_amdgcn_wmma_f32_16x16x32_f16(
                false, a.v, false, b.v, (short)0, acc[ti], false, false);
        }
    }

    // Epilogue: D layout (VGPR v, lane<16 -> M=v; lane>=16 -> M=8+v), add bias
#pragma unroll
    for (int ti = 0; ti < 7; ++ti) {
        int nt = nq + ti * 4;
        if (nt >= 25) continue;
        int col = nt * 16 + r0;
        int rb  = rowBase + mt * 16 + (sel << 3);
        float bcol = bias_s[col];
#pragma unroll
        for (int v = 0; v < 8; ++v) {
            int row = rb + v;
            if (row < T)
                xp[(long)row * G4 + col] = (_Float16)(acc[ti][v] + bcol);
        }
    }
}

// ---------------------------------------------------------------------------
// Kernel B: sequential LSTM scan + classifier. 1 block, 512 threads.
// Thread j<400 owns gate row j; its W_hh row lives in 100 VGPRs.
// ---------------------------------------------------------------------------
__device__ __forceinline__ float fsigmoid(float x) {
    return 1.f / (1.f + __expf(-x));
}
__device__ __forceinline__ float ftanh(float x) {
    float e = __expf(-2.f * x);
    return (1.f - e) / (1.f + e);
}

__global__ __launch_bounds__(512, 1) void lstm_scan_kernel(
    const _Float16* __restrict__ xp, const float* __restrict__ W_hh,
    const float* __restrict__ W1, const float* __restrict__ b1,
    const float* __restrict__ W2, const float* __restrict__ b2,
    float* __restrict__ out, int T)
{
    __shared__ __attribute__((aligned(16))) float h_s[128];
    __shared__ float gates_s[G4];
    __shared__ float hidden_s[64];

    const int j = threadIdx.x;

    // W_hh row j -> registers (fully unrolled so it stays in VGPRs)
    float w[H];
    if (j < G4) {
#pragma unroll
        for (int k = 0; k < H; ++k) w[k] = W_hh[j * H + k];
    }
    if (j < H) h_s[j] = 0.f;
    float c = 0.f;
    __syncthreads();

    // 8-deep register ring prefetch of this thread's x_proj column
    _Float16 xr[8];
    if (j < G4) {
#pragma unroll
        for (int s = 0; s < 8; ++s)
            xr[s] = (s < T) ? xp[(long)s * G4 + j] : (_Float16)0.f;
    }

    auto step = [&](float xv) {
        if (j < G4) {
            const float4* h4 = (const float4*)h_s;  // 25x ds_load_b128
            float a0 = 0.f, a1 = 0.f, a2 = 0.f, a3 = 0.f;
#pragma unroll
            for (int k4 = 0; k4 < H / 4; ++k4) {
                float4 hv = h4[k4];
                a0 = fmaf(hv.x, w[4 * k4 + 0], a0);
                a1 = fmaf(hv.y, w[4 * k4 + 1], a1);
                a2 = fmaf(hv.z, w[4 * k4 + 2], a2);
                a3 = fmaf(hv.w, w[4 * k4 + 3], a3);
            }
            gates_s[j] = xv + ((a0 + a1) + (a2 + a3));
        }
        __syncthreads();
        if (j < H) {
            float gi = fsigmoid(gates_s[j]);
            float gf = fsigmoid(gates_s[H + j]);
            float gg = ftanh(gates_s[2 * H + j]);
            float go = fsigmoid(gates_s[3 * H + j]);
            c = gf * c + gi * gg;
            h_s[j] = go * ftanh(c);
        }
        __syncthreads();
    };

    const int T8 = T & ~7;
    for (int t = 0; t < T8; t += 8) {
#pragma unroll
        for (int s = 0; s < 8; ++s) {
            float xv = (j < G4) ? (float)xr[s] : 0.f;
            int tn = t + 8 + s;                 // refill ring 8 steps ahead
            if (j < G4 && tn < T) xr[s] = xp[(long)tn * G4 + j];
            step(xv);
        }
    }
    for (int t = T8; t < T; ++t) {
        float xv = (j < G4) ? (float)xp[(long)t * G4 + j] : 0.f;
        step(xv);
    }

    // Classifier: hidden = relu(h @ W1.T + b1); pred = hidden @ W2.T + b2
    if (j < 64) {
        float acc = b1[j];
#pragma unroll
        for (int k = 0; k < H; ++k) acc = fmaf(h_s[k], W1[j * H + k], acc);
        hidden_s[j] = fmaxf(acc, 0.f);
    }
    __syncthreads();
    if (j < 7) {
        float acc = b2[j];
#pragma unroll
        for (int k = 0; k < 64; ++k) acc = fmaf(hidden_s[k], W2[j * 64 + k], acc);
        out[j] = acc;
    }
}

// ---------------------------------------------------------------------------
extern "C" void kernel_launch(void* const* d_in, const int* in_sizes, int n_in,
                              void* d_out, int out_size, void* d_ws, size_t ws_size,
                              hipStream_t stream)
{
    const float* x    = (const float*)d_in[0];
    const float* W_ih = (const float*)d_in[1];
    const float* W_hh = (const float*)d_in[2];
    const float* b_ih = (const float*)d_in[3];
    const float* b_hh = (const float*)d_in[4];
    const float* W1   = (const float*)d_in[5];
    const float* b1   = (const float*)d_in[6];
    const float* W2   = (const float*)d_in[7];
    const float* b2   = (const float*)d_in[8];
    float* out        = (float*)d_out;

    const int T = in_sizes[0] / H;          // 200000
    _Float16* xp = (_Float16*)d_ws;         // T*400 f16 = 160 MB (fits L2)
    _Float16* Bp = xp + (size_t)T * G4;     // 128 KB f16 B panel

    // Idempotent host-side attribute set (no stream op, no state guard).
    hipFuncSetAttribute((const void*)xproj_wmma_kernel,
                        hipFuncAttributeMaxDynamicSharedMemorySize,
                        SMEM_BYTES);

    prep_bpanel_kernel<<<1, 256, 0, stream>>>(W_ih, Bp);
    const int blocksA = (T + 31) / 32;
    xproj_wmma_kernel<<<blocksA, 256, SMEM_BYTES, stream>>>(
        x, Bp, b_ih, b_hh, xp, T);
    lstm_scan_kernel<<<1, 512, 0, stream>>>(xp, W_hh, W1, b1, W2, b2, out, T);
}
